// BasicBlock_37099927502865
// MI455X (gfx1250) — matrix operand order repaired and checked by reading, not compiled, not run on hardware
//
#include <hip/hip_runtime.h>

typedef __attribute__((ext_vector_type(16))) _Float16 v16h;
typedef __attribute__((ext_vector_type(8)))  _Float16 v8h;
typedef __attribute__((ext_vector_type(8)))  float    v8f;
typedef __attribute__((ext_vector_type(4)))  int      v4i;

#define CCH   256
#define HWSZ  3136              // 56*56
#define KTOT  2304              // 256*9
#define NPIX  100352            // 32*56*56
#define PADW  58                // 56 + halo
#define EPSB  1e-5f
#define NSTEP 72                // 9 taps * 8 ci-chunks of 32

#if __has_builtin(__builtin_amdgcn_global_load_async_to_lds_b128)
#define HAVE_ASYNC_LDS 1
#else
#define HAVE_ASYNC_LDS 0
#endif

typedef __attribute__((address_space(1))) v4i* gv4i_p;
typedef __attribute__((address_space(3))) v4i* lv4i_p;

#if HAVE_ASYNC_LDS
#define ASYNC_COPY(gp, lp) \
  __builtin_amdgcn_global_load_async_to_lds_b128((gv4i_p)(gp), (lv4i_p)(lp), 0, 0)
#if __has_builtin(__builtin_amdgcn_s_wait_asynccnt)
#define WAIT_ASYNC() __builtin_amdgcn_s_wait_asynccnt(0)
#else
#define WAIT_ASYNC() asm volatile("s_wait_asynccnt 0" ::: "memory")
#endif
#else
#define ASYNC_COPY(gp, lp) (*(v8h*)(lp) = *(const v8h*)(gp))
#define WAIT_ASYNC() ((void)0)
#endif

__device__ __forceinline__ float quantz(float v) {
  // round to nearest multiple of 1/128, ties toward upper (matches reference)
  return floorf(v * 128.0f + 0.5f) * 0.0078125f;
}

// ---------------------------------------------------------------------------
// Prep: binarize weights into fp16, layout wB[co][rs*256 + ci]; fold BN.
// ---------------------------------------------------------------------------
__global__ __launch_bounds__(256) void bb_prep(
    const float* __restrict__ w1, const float* __restrict__ w2,
    const float* __restrict__ g1, const float* __restrict__ b1,
    const float* __restrict__ m1, const float* __restrict__ v1,
    const float* __restrict__ g2, const float* __restrict__ b2,
    const float* __restrict__ m2, const float* __restrict__ v2,
    _Float16* __restrict__ wB1, _Float16* __restrict__ wB2,
    float* __restrict__ sc1, float* __restrict__ sh1,
    float* __restrict__ sc2, float* __restrict__ sh2)
{
  const int co = blockIdx.x;     // 0..255
  const int ci = threadIdx.x;    // 0..255
  const size_t inbase  = ((size_t)co * CCH + ci) * 9;   // OIHW flat
  const size_t outbase = (size_t)co * KTOT + ci;        // [co][rs*256+ci]
#pragma unroll
  for (int rs = 0; rs < 9; ++rs) {
    wB1[outbase + rs * 256] = (_Float16)(w1[inbase + rs] >= 0.0f ? 1.0f : -1.0f);
    wB2[outbase + rs * 256] = (_Float16)(w2[inbase + rs] >= 0.0f ? 1.0f : -1.0f);
  }
  if (co == 0) {
    const float i1 = g1[ci] * rsqrtf(v1[ci] + EPSB);
    sc1[ci] = i1;
    sh1[ci] = b1[ci] - m1[ci] * i1;
    const float i2 = g2[ci] * rsqrtf(v2[ci] + EPSB);
    sc2[ci] = i2;
    sh2[ci] = b2[ci] - m2[ci] * i2;
  }
}

// ---------------------------------------------------------------------------
// Zero the halo ring of both padded NHWC activation buffers (re-run per call:
// harness poisons the workspace).
// ---------------------------------------------------------------------------
__global__ __launch_bounds__(256) void bb_ring(_Float16* __restrict__ xq,
                                               _Float16* __restrict__ midp)
{
  const int b = blockIdx.x;            // 32 * 228 blocks
  const int n = b / 228;
  const int ridx = b - n * 228;
  int hh, ww;
  if (ridx < 58)       { hh = 0;          ww = ridx;       }
  else if (ridx < 116) { hh = 57;         ww = ridx - 58;  }
  else if (ridx < 172) { hh = ridx - 115; ww = 0;          }
  else                 { hh = ridx - 171; ww = 57;         }
  const size_t base = ((size_t)(n * PADW + hh) * PADW + ww) * CCH + threadIdx.x;
  xq[base]   = (_Float16)0.0f;
  midp[base] = (_Float16)0.0f;
}

// ---------------------------------------------------------------------------
// Quantize x once and transpose NCHW fp32 -> padded NHWC fp16 (via LDS so
// both global sides are coalesced).
// ---------------------------------------------------------------------------
__global__ __launch_bounds__(256) void bb_quant(const float* __restrict__ x,
                                                _Float16* __restrict__ xq)
{
  __shared__ _Float16 tile[32 * 56];
  const int n = blockIdx.x / 56;       // 32*56 blocks, one (n, h) row each
  const int h = blockIdx.x - n * 56;
  const int tid = threadIdx.x;
  for (int cblk = 0; cblk < 8; ++cblk) {
    __syncthreads();
#pragma unroll
    for (int it = 0; it < 7; ++it) {   // 7*256 = 1792 = 32*56
      const int e = it * 256 + tid;
      const int c = e / 56;
      const int w = e - c * 56;
      const float v = x[((size_t)(n * CCH + cblk * 32 + c) * 56 + h) * 56 + w];
      tile[c * 56 + w] = (_Float16)quantz(v);
    }
    __syncthreads();
#pragma unroll
    for (int it = 0; it < 7; ++it) {
      const int e = it * 256 + tid;
      const int w = e >> 5;
      const int c = e & 31;
      xq[((size_t)(n * PADW + h + 1) * PADW + (w + 1)) * CCH + cblk * 32 + c] =
          tile[c * 56 + w];
    }
  }
}

// ---------------------------------------------------------------------------
// Shared implicit-GEMM core: block = 64 pixels x 256 couts, 8 wave32,
// wave owns a 64x32 C slab (8 accumulator tiles).
// Software-pipelined: double-buffered async-to-LDS A staging overlaps the
// WMMAs; B fragments are register-pipelined one step ahead.
// ---------------------------------------------------------------------------
__device__ __forceinline__ void conv_core(
    const _Float16* __restrict__ src,   // padded NHWC, quantized fp16
    const _Float16* __restrict__ wB,
    v8f acc[4][2])
{
  __shared__ alignas(16) _Float16 aT[2][64 * 32];

  const int tid  = threadIdx.x;
  const int lane = tid & 31;
  const int wv   = tid >> 5;        // 0..7
  const int lhi  = lane >> 4;
  const int llo  = lane & 15;
  const int n0   = wv * 32;

  const int pbase = blockIdx.x * 64;
  const int pl    = tid >> 2;       // staging pixel 0..63
  const int ch    = tid & 3;        // 16B chunk within that pixel's 64B row
  const int p  = pbase + pl;
  const int n  = p / HWSZ;
  const int hw = p - n * HWSZ;
  const int h  = hw / 56;
  const int w  = hw - h * 56;

  // Staging source base for this thread (tap offset added per step; padded
  // coords absorb the -1 halo shift).
  const _Float16* base0 =
      src + ((size_t)(n * PADW + h) * PADW + w) * CCH + ch * 8;
  _Float16* ldst = &aT[0][pl * 32 + ch * 8];

  // B bases: kk(s) = rs*256 + cb*32 + lhi*16 = s*32 + lhi*16 (linear in s).
  const _Float16* b0base = wB + (size_t)(n0 + llo)      * KTOT + lhi * 16;
  const _Float16* b1base = wB + (size_t)(n0 + 16 + llo) * KTOT + lhi * 16;

  // Issue async A-tile copy for step sn into buffer sn&1.
#define ISSUE(sn) do {                                                       \
    const int rsn_  = (sn) >> 3;                                             \
    const int offn_ = rsn_ + (rsn_ / 3) * 55;  /* (r*PADW + s) tap offset */ \
    ASYNC_COPY(base0 + (size_t)offn_ * CCH + ((sn) & 7) * 32,                \
               ldst + ((sn) & 1) * (64 * 32));                               \
  } while (0)

  ISSUE(0);
  v16h bf0 = *(const v16h*)(b0base);
  v16h bf1 = *(const v16h*)(b1base);

#pragma unroll 8
  for (int s = 0; s < NSTEP; ++s) {
    WAIT_ASYNC();                   // my part of tile s is in LDS
    __syncthreads();                // everyone's part is; compute s-1 done
    if (s + 1 < NSTEP) ISSUE(s + 1);
    const int sn = (s + 1 < NSTEP) ? s + 1 : NSTEP - 1;
    const v16h nbf0 = *(const v16h*)(b0base + sn * 32);
    const v16h nbf1 = *(const v16h*)(b1base + sn * 32);
    __builtin_prefetch(b0base + sn * 32 + 32, 0, 0);

    const _Float16* at = &aT[s & 1][0];
    v16h af[4];
#pragma unroll
    for (int mt = 0; mt < 4; ++mt) {
      const int m = mt * 16 + llo;
      ((v8h*)&af[mt])[0] = *(const v8h*)(at + m * 32 + lhi * 16);     // K = 16*lhi + 0..7
      ((v8h*)&af[mt])[1] = *(const v8h*)(at + m * 32 + lhi * 16 + 8); // K = 16*lhi + 8..15
    }
#pragma unroll
    for (int mt = 0; mt < 4; ++mt) {
      acc[mt][0] = __builtin_amdgcn_wmma_f32_16x16x32_f16(
          false, af[mt], false, bf0, (short)0, acc[mt][0], false, false);
      acc[mt][1] = __builtin_amdgcn_wmma_f32_16x16x32_f16(
          false, af[mt], false, bf1, (short)0, acc[mt][1], false, false);
    }
    bf0 = nbf0;
    bf1 = nbf1;
  }
#undef ISSUE
}

// ---------------------------------------------------------------------------
// Conv1: xq (*) sign(w1) -> BN1 -> hardtanh -> quantize -> padded NHWC fp16.
// ---------------------------------------------------------------------------
__global__ __launch_bounds__(256) void bb_conv1(
    const _Float16* __restrict__ xq, const _Float16* __restrict__ wB,
    const float* __restrict__ scale, const float* __restrict__ shift,
    _Float16* __restrict__ midp)
{
  v8f acc[4][2] = {};
  conv_core(xq, wB, acc);

  const int tid  = threadIdx.x;
  const int lane = tid & 31;
  const int wv   = tid >> 5;
  const int lhi  = lane >> 4;
  const int llo  = lane & 15;
  const int n0   = wv * 32;
  const int pbase = blockIdx.x * 64;

#pragma unroll
  for (int mt = 0; mt < 4; ++mt) {
#pragma unroll
    for (int nt = 0; nt < 2; ++nt) {
      const int co = n0 + nt * 16 + llo;
      const float sc = scale[co];
      const float sh = shift[co];
#pragma unroll
      for (int i = 0; i < 8; ++i) {
        const int pp  = pbase + mt * 16 + lhi * 8 + i;   // C layout: M = i + 8*lhi
        const int nn  = pp / HWSZ;
        const int hw2 = pp - nn * HWSZ;
        const int hh  = hw2 / 56;
        const int ww  = hw2 - hh * 56;
        float v = acc[mt][nt][i] * sc + sh;
        v = fminf(1.0f, fmaxf(-1.0f, v));
        v = quantz(v);
        midp[((size_t)(nn * PADW + hh + 1) * PADW + (ww + 1)) * CCH + co] =
            (_Float16)v;
      }
    }
  }
}

// ---------------------------------------------------------------------------
// Conv2: midp (*) sign(w2) + residual -> BN2 -> hardtanh -> fp32 NCHW out.
// ---------------------------------------------------------------------------
__global__ __launch_bounds__(256) void bb_conv2(
    const _Float16* __restrict__ midp, const _Float16* __restrict__ wB,
    const float* __restrict__ scale, const float* __restrict__ shift,
    const float* __restrict__ xres, float* __restrict__ out)
{
  v8f acc[4][2] = {};
  conv_core(midp, wB, acc);

  const int tid  = threadIdx.x;
  const int lane = tid & 31;
  const int wv   = tid >> 5;
  const int lhi  = lane >> 4;
  const int llo  = lane & 15;
  const int n0   = wv * 32;
  const int pbase = blockIdx.x * 64;

#pragma unroll
  for (int mt = 0; mt < 4; ++mt) {
#pragma unroll
    for (int nt = 0; nt < 2; ++nt) {
      const int co = n0 + nt * 16 + llo;
      const float sc = scale[co];
      const float sh = shift[co];
#pragma unroll
      for (int i = 0; i < 8; ++i) {
        const int pp  = pbase + mt * 16 + lhi * 8 + i;
        const int nn  = pp / HWSZ;
        const int hw2 = pp - nn * HWSZ;
        const size_t idx = (size_t)(nn * CCH + co) * HWSZ + hw2;
        float v = acc[mt][nt][i] + xres[idx];
        v = v * sc + sh;
        v = fminf(1.0f, fmaxf(-1.0f, v));
        out[idx] = v;
      }
    }
  }
}

// ---------------------------------------------------------------------------
extern "C" void kernel_launch(void* const* d_in, const int* in_sizes, int n_in,
                              void* d_out, int out_size, void* d_ws, size_t ws_size,
                              hipStream_t stream) {
  const float* x  = (const float*)d_in[0];
  const float* w1 = (const float*)d_in[1];
  const float* w2 = (const float*)d_in[2];
  const float* g1 = (const float*)d_in[3];
  const float* b1 = (const float*)d_in[4];
  const float* m1 = (const float*)d_in[5];
  const float* v1 = (const float*)d_in[6];
  const float* g2 = (const float*)d_in[7];
  const float* b2 = (const float*)d_in[8];
  const float* m2 = (const float*)d_in[9];
  const float* v2 = (const float*)d_in[10];
  float* out = (float*)d_out;

  // Workspace layout
  char* ws = (char*)d_ws;
  _Float16* wB1  = (_Float16*)ws;                      // 1,179,648 B
  _Float16* wB2  = (_Float16*)(ws + 1179648);          // 1,179,648 B
  float*    sc1  = (float*)(ws + 2359296);             // 4096 B total
  float*    sh1  = sc1 + 256;
  float*    sc2  = sc1 + 512;
  float*    sh2  = sc1 + 768;
  _Float16* xq   = (_Float16*)(ws + 2363392);          // 32*58*58*256*2 = 55,115,776 B
  _Float16* midp = (_Float16*)(ws + 2363392 + 55115776);

  bb_prep<<<256, 256, 0, stream>>>(w1, w2, g1, b1, m1, v1, g2, b2, m2, v2,
                                   wB1, wB2, sc1, sh1, sc2, sh2);
  bb_ring<<<32 * 228, 256, 0, stream>>>(xq, midp);
  bb_quant<<<32 * 56, 256, 0, stream>>>(x, xq);
  bb_conv1<<<NPIX / 64, 256, 0, stream>>>(xq, wB1, sc1, sh1, midp);
  bb_conv2<<<NPIX / 64, 256, 0, stream>>>(midp, wB2, sc2, sh2, x, out);
}
